// TAGNNetwork_15307263443521
// MI455X (gfx1250) — matrix operand order repaired and checked by reading, not compiled, not run on hardware
//
#include <hip/hip_runtime.h>

#define NNODES 1024
#define DFEAT  64
#define SFLOWS 16384
#define BATCH  4

typedef __attribute__((ext_vector_type(2)))  float  v2f;
typedef __attribute__((ext_vector_type(8)))  float  v8f;
typedef __attribute__((ext_vector_type(16))) __bf16 v16bf;

static __device__ __forceinline__ unsigned pack_bf16(float lo, float hi) {
  // round-to-nearest-even f32 -> bf16, packed {hi,lo}
  unsigned a = __float_as_uint(lo); a += 0x7fffu + ((a >> 16) & 1u);
  unsigned b = __float_as_uint(hi); b += 0x7fffu + ((b >> 16) & 1u);
  return (a >> 16) | (b & 0xffff0000u);
}

// ---------------- K0: zero accumulation buffers ----------------
__global__ void k_zero(float* __restrict__ node, unsigned* __restrict__ mask) {
  int tid = blockIdx.x * 256 + threadIdx.x;          // 262144 threads
  node[tid] = 0.0f;                                  // B*N*D floats
  if (tid < BATCH * NNODES * 32) mask[tid] = 0u;     // B*N*N bits
}

// ---------------- K1: scatter-add flow features ----------------
__global__ void k_scatter(const float* __restrict__ flow,
                          const int* __restrict__ dst,
                          float* __restrict__ node) {
  int tid = blockIdx.x * 256 + threadIdx.x;          // B*S*D = 4194304
  int d  = tid & (DFEAT - 1);
  int fs = tid >> 6;                                 // b*S + s
  int b  = fs >> 14;                                 // S = 16384
  int n  = dst[fs];
  atomicAdd(&node[((b * NNODES + n) << 6) + d], flow[tid]);
}

// ---------------- K2: edge existence bitmask (adj>0 <=> edge) ----------------
__global__ void k_edges(const int* __restrict__ src, const int* __restrict__ dst,
                        unsigned* __restrict__ mask) {
  int tid = blockIdx.x * 256 + threadIdx.x;          // B*S = 65536
  int b = tid >> 14;
  int i = src[tid], j = dst[tid];
  unsigned* mb = mask + b * (NNODES * 32);
  atomicOr(&mb[i * 32 + (j >> 5)], 1u << (j & 31));
  atomicOr(&mb[j * 32 + (i >> 5)], 1u << (i & 31));
}

// ---------------- K3: row L2-normalize (in place) ----------------
__global__ __launch_bounds__(32) void k_normalize(float* __restrict__ node) {
  int row = blockIdx.x, lane = threadIdx.x;          // 4096 rows
  float2* p = (float2*)node + row * 32 + lane;
  float2 v = *p;
  float ss = v.x * v.x + v.y * v.y;
  for (int o = 16; o > 0; o >>= 1) ss += __shfl_xor(ss, o);
  float s = 1.0f / fmaxf(sqrtf(ss), 1e-12f);
  *p = make_float2(v.x * s, v.y * s);
}

// ---------------- K4: Wh = h @ W via f32 WMMA; emit f32 Wh + bf16 swizzled B-operand ----------------
__global__ __launch_bounds__(128) void k_wh(const float* __restrict__ h,
                                            const float* __restrict__ W,
                                            float* __restrict__ Wh,
                                            unsigned* __restrict__ WhB) {
  __shared__ float tile[4][16][17];
  int wave = threadIdx.x >> 5, lane = threadIdx.x & 31;
  int it = blockIdx.x;                               // 0..255 global 16-row tiles
  int rowbase = it << 4;
  int nt = wave;                                     // 16-col tile of W
  int m = lane & 15, halfk = lane >> 4;

  v8f c = {};
  for (int kt = 0; kt < 16; ++kt) {                  // K=64, 4 per WMMA
    int k0 = kt * 4 + halfk * 2;
    v2f a, bf;
    float2 hv = ((const float2*)h)[(rowbase + m) * 32 + (k0 >> 1)];
    a.x = hv.x; a.y = hv.y;                          // A: 16x4, lane half = K-pair
    bf.x = W[k0 * DFEAT + nt * 16 + m];              // B: 4x16, lane = column
    bf.y = W[(k0 + 1) * DFEAT + nt * 16 + m];
    c = __builtin_amdgcn_wmma_f32_16x16x4_f32(false, a, false, bf,
                                              (short)0, c, false, false);
  }
  for (int r = 0; r < 8; ++r) {                      // C/D: M = r + half*8, N = lane&15
    int mm = r + halfk * 8;
    Wh[(rowbase + mm) * DFEAT + nt * 16 + m] = c[r];
    tile[wave][mm][m] = c[r];
  }
  __syncthreads();

  // Re-emit tile as WMMA-B-operand bf16 fragments: [b][kt(32 K)][nt][lane][8 dwords]
  int b = it >> 6, itl = it & 63;
  int kt2 = itl >> 1, half = itl & 1;                // this tile = one 16-row K-half
  int vb = halfk * 4, n = m;
  uint4 pk;
  pk.x = pack_bf16(tile[wave][2 * vb + 0][n], tile[wave][2 * vb + 1][n]);
  pk.y = pack_bf16(tile[wave][2 * vb + 2][n], tile[wave][2 * vb + 3][n]);
  pk.z = pack_bf16(tile[wave][2 * vb + 4][n], tile[wave][2 * vb + 5][n]);
  pk.w = pack_bf16(tile[wave][2 * vb + 6][n], tile[wave][2 * vb + 7][n]);
  unsigned* dstp = WhB + ((((b * 32 + kt2) * 4 + nt) * 32) + (half * 16 + n)) * 8 + vb;
  *(uint4*)dstp = pk;
}

// ---------------- K5: f1 = Wh@a1, f2 = Wh@a2 ----------------
__global__ __launch_bounds__(128) void k_f12(const float* __restrict__ Wh,
                                             const float* __restrict__ a1,
                                             const float* __restrict__ a2,
                                             float* __restrict__ f1,
                                             float* __restrict__ f2) {
  int wave = threadIdx.x >> 5, lane = threadIdx.x & 31;
  int row = blockIdx.x * 4 + wave;                   // 4096 rows
  float2 w = ((const float2*)Wh)[row * 32 + lane];
  float2 x = ((const float2*)a1)[lane];
  float2 y = ((const float2*)a2)[lane];
  float s1 = w.x * x.x + w.y * x.y;
  float s2 = w.x * y.x + w.y * y.y;
  for (int o = 16; o > 0; o >>= 1) { s1 += __shfl_xor(s1, o); s2 += __shfl_xor(s2, o); }
  if (lane == 0) { f1[row] = s1; f2[row] = s2; }
}

// ---------------- K6: fused masked softmax + P@V (bf16 WMMA) + ELU ----------------
__global__ __launch_bounds__(32) void k_attn(const float* __restrict__ f1,
                                             const float* __restrict__ f2,
                                             const unsigned* __restrict__ mask,
                                             const unsigned* __restrict__ WhB,
                                             float* __restrict__ out) {
  __shared__ float    f2s[NNODES];
  __shared__ unsigned mrow[16 * 32];
  __shared__ float    f1s[16], rmax[16], rsum[16];
  int lane = threadIdx.x;
  int it = blockIdx.x, b = blockIdx.y;               // 64 row-tiles x 4 batches

  const unsigned* mb = mask + b * (NNODES * 32) + (it * 16) * 32;
  for (int k = 0; k < 32; ++k) f2s[k * 32 + lane] = f2[b * NNODES + k * 32 + lane];
  for (int k = 0; k < 16; ++k) mrow[k * 32 + lane] = mb[k * 32 + lane];
  if (lane < 16) f1s[lane] = f1[b * NNODES + it * 16 + lane];
  __syncthreads();

  // Pass 1: per-row max and sum of exp over masked leaky_relu(f1_i + f2_j)
  for (int r = 0; r < 16; ++r) {
    float f1r = f1s[r];
    float mx = -1e30f;
    for (int k = 0; k < 32; ++k)
      if ((mrow[r * 32 + k] >> lane) & 1u) {
        float e = f1r + f2s[k * 32 + lane];
        e = e > 0.0f ? e : 0.2f * e;
        mx = fmaxf(mx, e);
      }
    for (int o = 16; o > 0; o >>= 1) mx = fmaxf(mx, __shfl_xor(mx, o));
    float sm = 0.0f;
    for (int k = 0; k < 32; ++k)
      if ((mrow[r * 32 + k] >> lane) & 1u) {
        float e = f1r + f2s[k * 32 + lane];
        e = e > 0.0f ? e : 0.2f * e;
        sm += __expf(e - mx);
      }
    for (int o = 16; o > 0; o >>= 1) sm += __shfl_xor(sm, o);
    if (lane == 0) { rmax[r] = mx; rsum[r] = sm; }
  }
  __syncthreads();

  // Pass 2: out(16x64) = P(16x1024) @ WhB(1024x64) in K-tiles of 32 via bf16 WMMA
  int m = lane & 15, halfsel = lane >> 4;
  float f1r = f1s[m], rm = rmax[m], rs = rsum[m];
  bool uni = (rs == 0.0f);                           // isolated row -> uniform softmax
  float ri = uni ? 0.0f : 1.0f / rs;

  v8f acc[4] = {v8f{}, v8f{}, v8f{}, v8f{}};
  const uint4* bbase = (const uint4*)WhB;
  for (int kt = 0; kt < 32; ++kt) {
    unsigned au[8];
    for (int v = 0; v < 8; ++v) {
      float pv[2];
      for (int hh = 0; hh < 2; ++hh) {
        int k = (halfsel << 4) + (v << 1) + hh;  // A 16x32 bf16 layout
        int j = kt * 32 + k;
        float p;
        if (uni) p = 0.0009765625f;                  // 1/1024
        else if ((mrow[m * 32 + (j >> 5)] >> (j & 31)) & 1u) {
          float e = f1r + f2s[j];
          e = e > 0.0f ? e : 0.2f * e;
          p = __expf(e - rm) * ri;
        } else p = 0.0f;
        pv[hh] = p;
      }
      au[v] = pack_bf16(pv[0], pv[1]);
    }
    v16bf afrag;
    { union { unsigned u[8]; v16bf v; } cv;
      for (int v = 0; v < 8; ++v) cv.u[v] = au[v];
      afrag = cv.v; }

    for (int nt = 0; nt < 4; ++nt) {
      int idx = ((b * 32 + kt) * 4 + nt) * 64 + lane * 2;
      union { uint4 q[2]; v16bf v; } bw;
      bw.q[0] = bbase[idx];
      bw.q[1] = bbase[idx + 1];
      acc[nt] = __builtin_amdgcn_wmma_f32_16x16x32_bf16(false, afrag, false, bw.v,
                                                        (short)0, acc[nt], false, false);
    }
  }

  // ELU epilogue
  for (int nt = 0; nt < 4; ++nt)
    for (int r = 0; r < 8; ++r) {
      float x = acc[nt][r];
      float y = x > 0.0f ? x : (__expf(x) - 1.0f);
      int i = it * 16 + r + halfsel * 8;
      int d = nt * 16 + m;
      out[((b * NNODES + i) << 6) + d] = y;
    }
}

extern "C" void kernel_launch(void* const* d_in, const int* in_sizes, int n_in,
                              void* d_out, int out_size, void* d_ws, size_t ws_size,
                              hipStream_t stream) {
  (void)in_sizes; (void)n_in; (void)out_size; (void)ws_size;
  const float* flow = (const float*)d_in[0];
  const int*   src  = (const int*)d_in[1];
  const int*   dst  = (const int*)d_in[2];
  // d_in[3..8] (flow_volumes, emb, We1, be1, We2, be2) are provably dead:
  // w = sigmoid(.)*sigmoid(.) > 0 always, so adj>0 reduces to edge existence.
  const float* W  = (const float*)d_in[9];
  const float* a1 = (const float*)d_in[10];
  const float* a2 = (const float*)d_in[11];
  float* out = (float*)d_out;

  char* ws = (char*)d_ws;
  float*    node = (float*)ws;                                   // 1 MB (h in place)
  float*    Wh   = (float*)(ws + (1 << 20));                     // 1 MB
  unsigned* WhB  = (unsigned*)(ws + (2 << 20));                  // 512 KB bf16 B-operand
  float*    f1   = (float*)(ws + (2 << 20) + (512 << 10));       // 16 KB
  float*    f2   = f1 + BATCH * NNODES;                          // 16 KB
  unsigned* mask = (unsigned*)(ws + (2 << 20) + (512 << 10) + (32 << 10)); // 512 KB

  k_zero     <<<1024, 256, 0, stream>>>(node, mask);
  k_scatter  <<<16384, 256, 0, stream>>>(flow, dst, node);
  k_edges    <<<256, 256, 0, stream>>>(src, dst, mask);
  k_normalize<<<4096, 32, 0, stream>>>(node);
  k_wh       <<<256, 128, 0, stream>>>(node, W, Wh, WhB);
  k_f12      <<<1024, 128, 0, stream>>>(Wh, a1, a2, f1, f2);
  k_attn     <<<dim3(64, 4), 32, 0, stream>>>(f1, f2, mask, WhB, out);
}
